// DecoderRNN_78383153152469
// MI455X (gfx1250) — matrix operand order repaired and checked by reading, not compiled, not run on hardware
//
#include <hip/hip_runtime.h>
#include <hip/hip_bf16.h>
#include <math.h>

// ---------------------------------------------------------------------------
// DecoderRNN single step on MI455X (gfx1250, wave32).
// Bandwidth-bound: ~243 MB of weight streaming per call -> ~10.4 us floor at
// 23.3 TB/s. The 205 MB out_w projection is done with v_wmma_f32_16x16x32_f16
// (h replicated across A rows, 16 out_w rows per wave as B columns), fed by
// non-temporal 128-bit loads so the reusable small weights stay in L2.
// ---------------------------------------------------------------------------

#define VSZ 50257
#define HSZ 1024
#define ESZ 1024
#define LSZ 512

typedef __attribute__((ext_vector_type(16))) _Float16 v16h;
typedef __attribute__((ext_vector_type(8)))  float    v8f;
typedef __attribute__((ext_vector_type(4)))  float    f32x4;

__device__ __forceinline__ float wave_sum(float v) {
#pragma unroll
  for (int off = 16; off > 0; off >>= 1) v += __shfl_xor(v, off, 32);
  return v;
}

__device__ __forceinline__ float sigmoidf_(float x) { return 1.0f / (1.0f + expf(-x)); }

// ---------------- k1: attn logits = [emb, h] @ attn_w.T + attn_b -----------
__global__ void k1_attn_logits(const int* __restrict__ tok,
                               const float* __restrict__ hid,
                               const float* __restrict__ embedding,
                               const float* __restrict__ attn_w,
                               const float* __restrict__ attn_b,
                               float* __restrict__ attn_logits) {
  const int lane = threadIdx.x & 31;
  const int row  = blockIdx.x * (blockDim.x >> 5) + (threadIdx.x >> 5);
  if (row >= LSZ) return;
  const float* emb = embedding + (size_t)tok[0] * ESZ;
  const float* wr  = attn_w + (size_t)row * (HSZ + ESZ);
  float acc = 0.0f;
  for (int j = lane * 4; j < HSZ + ESZ; j += 128) {
    f32x4 wv = *(const f32x4*)(wr + j);
    f32x4 xv = (j < ESZ) ? *(const f32x4*)(emb + j)
                         : *(const f32x4*)(hid + (j - ESZ));
#pragma unroll
    for (int i = 0; i < 4; ++i) acc += wv[i] * xv[i];
  }
  acc = wave_sum(acc);
  if (lane == 0) attn_logits[row] = acc + attn_b[row];
}

// ---------------- k2: softmax over 512 logits ------------------------------
__global__ void k2_softmax(const float* __restrict__ attn_logits,
                           float* __restrict__ attn_wt,
                           float* __restrict__ out_attn) {
  __shared__ float red[LSZ];
  const int t = threadIdx.x;
  float v = attn_logits[t];
  red[t] = v; __syncthreads();
  for (int s = LSZ / 2; s > 0; s >>= 1) {
    if (t < s) red[t] = fmaxf(red[t], red[t + s]);
    __syncthreads();
  }
  const float m = red[0]; __syncthreads();
  const float e = expf(v - m);
  red[t] = e; __syncthreads();
  for (int s = LSZ / 2; s > 0; s >>= 1) {
    if (t < s) red[t] += red[t + s];
    __syncthreads();
  }
  const float w = e / red[0];
  attn_wt[t] = w;
  out_attn[t] = w;
}

// ---------------- k3: attn_applied = w @ encoder_outputs -------------------
__global__ void k3_attn_applied(const float* __restrict__ attn_wt,
                                const float* __restrict__ enc,
                                float* __restrict__ attn_app) {
  __shared__ float sw[LSZ];
  for (int i = threadIdx.x; i < LSZ; i += blockDim.x) sw[i] = attn_wt[i];
  __syncthreads();
  const int j = blockIdx.x * blockDim.x + threadIdx.x;  // 0..1023
  float acc = 0.0f;
  for (int l = 0; l < LSZ; ++l) acc += sw[l] * enc[(size_t)l * HSZ + j];
  attn_app[j] = acc;
}

// ---------------- k4: x = relu([emb, attn_app] @ comb_w.T + comb_b) --------
__global__ void k4_combine(const int* __restrict__ tok,
                           const float* __restrict__ embedding,
                           const float* __restrict__ attn_app,
                           const float* __restrict__ comb_w,
                           const float* __restrict__ comb_b,
                           float* __restrict__ xout) {
  const int lane = threadIdx.x & 31;
  const int row  = blockIdx.x * (blockDim.x >> 5) + (threadIdx.x >> 5);
  if (row >= ESZ) return;
  const float* emb = embedding + (size_t)tok[0] * ESZ;
  const float* wr  = comb_w + (size_t)row * (ESZ + HSZ);
  float acc = 0.0f;
  for (int j = lane * 4; j < ESZ + HSZ; j += 128) {
    f32x4 wv = *(const f32x4*)(wr + j);
    f32x4 xv = (j < ESZ) ? *(const f32x4*)(emb + j)
                         : *(const f32x4*)(attn_app + (j - ESZ));
#pragma unroll
    for (int i = 0; i < 4; ++i) acc += wv[i] * xv[i];
  }
  acc = wave_sum(acc);
  if (lane == 0) xout[row] = fmaxf(acc + comb_b[row], 0.0f);
}

// ---------------- k5: gi = w_ih@x + b_ih ; gh = w_hh@h + b_hh --------------
__global__ void k5_gates(const float* __restrict__ x,
                         const float* __restrict__ hid,
                         const float* __restrict__ w_ih,
                         const float* __restrict__ w_hh,
                         const float* __restrict__ b_ih,
                         const float* __restrict__ b_hh,
                         float* __restrict__ gi,
                         float* __restrict__ gh) {
  const int lane = threadIdx.x & 31;
  const int row  = blockIdx.x * (blockDim.x >> 5) + (threadIdx.x >> 5);
  if (row >= 3 * HSZ) return;
  const float* wi = w_ih + (size_t)row * ESZ;
  const float* wh = w_hh + (size_t)row * HSZ;
  float ai = 0.0f, ah = 0.0f;
  for (int j = lane * 4; j < HSZ; j += 128) {
    f32x4 a = *(const f32x4*)(wi + j);
    f32x4 xv = *(const f32x4*)(x + j);
    f32x4 b = *(const f32x4*)(wh + j);
    f32x4 hv = *(const f32x4*)(hid + j);
#pragma unroll
    for (int i = 0; i < 4; ++i) { ai += a[i] * xv[i]; ah += b[i] * hv[i]; }
  }
  ai = wave_sum(ai);
  ah = wave_sum(ah);
  if (lane == 0) {
    gi[row] = ai + b_ih[row];
    gh[row] = ah + b_hh[row];
  }
}

// ---------------- k6: GRU gate math -> h_new (f32 + f16 copy) --------------
__global__ void k6_hnew(const float* __restrict__ gi,
                        const float* __restrict__ gh,
                        const float* __restrict__ hid,
                        float* __restrict__ hnew_ws,
                        float* __restrict__ hnew_out,
                        _Float16* __restrict__ h16) {
  const int i = blockIdx.x * blockDim.x + threadIdx.x;  // 0..1023
  const float r = sigmoidf_(gi[i] + gh[i]);
  const float z = sigmoidf_(gi[HSZ + i] + gh[HSZ + i]);
  const float n = tanhf(gi[2 * HSZ + i] + r * gh[2 * HSZ + i]);
  const float h = (1.0f - z) * n + z * hid[i];
  hnew_ws[i]  = h;
  hnew_out[i] = h;
  h16[i] = (_Float16)h;
}

// ---------------- k7: logits = h_new @ out_w.T + out_b (WMMA) --------------
// Each wave produces 16 logits. A = h (f16) replicated on all 16 rows, read
// from LDS in contiguous k order (k = 16h + s, as B); B = 16 out_w rows as columns, loaded with
// non-temporal b128 loads and converted to f16. f32 accumulate via
// v_wmma_f32_16x16x32_f16, K advances by 32 per issue.
__global__ void k7_out_gemv(const float* __restrict__ out_w,
                            const float* __restrict__ out_b,
                            const _Float16* __restrict__ h16,
                            float* __restrict__ logits) {
  __shared__ _Float16 sh[HSZ];
  for (int i = threadIdx.x; i < HSZ; i += blockDim.x) sh[i] = h16[i];
  __syncthreads();

  const int lane = threadIdx.x & 31;
  const int wave = blockIdx.x * (blockDim.x >> 5) + (threadIdx.x >> 5);
  const int r0   = wave * 16;
  const int n    = lane & 15;
  const bool hi  = lane >= 16;
  const int koff = hi ? 16 : 0;  // B half-wave K offset
  const int aoff = hi ? 16 : 0;  // A half-wave K offset

  int row = r0 + n;
  int rclamp = row < VSZ ? row : (VSZ - 1);
  const float* wrow = out_w + (size_t)rclamp * HSZ;

  v8f c = {};
  for (int k0 = 0; k0 < HSZ; k0 += 32) {
    // A fragment: all 16 M-rows hold the same h chunk.
    v16h a;
#pragma unroll
    for (int i = 0; i < 8; ++i) {
      a[i]     = sh[k0 + aoff + i];
      a[8 + i] = sh[k0 + aoff + 8 + i];
    }
    // B fragment: column n <- out_w row (r0+n), K = k0+koff .. +15.
    const f32x4* p = (const f32x4*)(wrow + k0 + koff);
    f32x4 f0 = __builtin_nontemporal_load(p + 0);
    f32x4 f1 = __builtin_nontemporal_load(p + 1);
    f32x4 f2 = __builtin_nontemporal_load(p + 2);
    f32x4 f3 = __builtin_nontemporal_load(p + 3);
    v16h b;
#pragma unroll
    for (int i = 0; i < 4; ++i) {
      b[i]      = (_Float16)f0[i];
      b[4 + i]  = (_Float16)f1[i];
      b[8 + i]  = (_Float16)f2[i];
      b[12 + i] = (_Float16)f3[i];
    }
    c = __builtin_amdgcn_wmma_f32_16x16x32_f16(
        /*neg_a=*/false, a, /*neg_b=*/false, b,
        /*c_mod=*/(short)0, c, /*reuse_a=*/false, /*reuse_b=*/false);
  }
  // All M rows identical; lanes 0..15 hold N=lane in c[0].
  if (!hi && row < VSZ) logits[row] = c[0] + out_b[row];
}

// ---------------- k8: max + log-sum-exp over 50257 logits ------------------
__global__ void k8_lse(const float* __restrict__ logits, float* __restrict__ red2) {
  __shared__ float red[256];
  const int t = threadIdx.x;
  float m = -INFINITY;
  for (int i = t; i < VSZ; i += 256) m = fmaxf(m, logits[i]);
  red[t] = m; __syncthreads();
  for (int s = 128; s > 0; s >>= 1) {
    if (t < s) red[t] = fmaxf(red[t], red[t + s]);
    __syncthreads();
  }
  m = red[0]; __syncthreads();
  float sum = 0.0f;
  for (int i = t; i < VSZ; i += 256) sum += expf(logits[i] - m);
  red[t] = sum; __syncthreads();
  for (int s = 128; s > 0; s >>= 1) {
    if (t < s) red[t] += red[t + s];
    __syncthreads();
  }
  if (t == 0) { red2[0] = m; red2[1] = logf(red[0]); }
}

// ---------------- k9: out = logits - max - lse -----------------------------
__global__ void k9_logsoftmax(const float* __restrict__ logits,
                              const float* __restrict__ red2,
                              float* __restrict__ out) {
  const int i = blockIdx.x * blockDim.x + threadIdx.x;
  if (i < VSZ) out[i] = logits[i] - red2[0] - red2[1];
}

// ---------------------------------------------------------------------------
extern "C" void kernel_launch(void* const* d_in, const int* in_sizes, int n_in,
                              void* d_out, int out_size, void* d_ws, size_t ws_size,
                              hipStream_t stream) {
  const int*   tok      = (const int*)d_in[0];
  const float* hid      = (const float*)d_in[1];
  const float* enc      = (const float*)d_in[2];
  const float* emb      = (const float*)d_in[3];
  const float* attn_w   = (const float*)d_in[4];
  const float* attn_b   = (const float*)d_in[5];
  const float* comb_w   = (const float*)d_in[6];
  const float* comb_b   = (const float*)d_in[7];
  const float* w_ih     = (const float*)d_in[8];
  const float* w_hh     = (const float*)d_in[9];
  const float* b_ih     = (const float*)d_in[10];
  const float* b_hh     = (const float*)d_in[11];
  const float* out_w    = (const float*)d_in[12];
  const float* out_b    = (const float*)d_in[13];

  float* out       = (float*)d_out;          // [0, V): log-probs
  float* out_hnew  = out + VSZ;              // [V, V+H): h_new
  float* out_attn  = out + VSZ + HSZ;        // [V+H, V+H+L): attn weights

  float* ws          = (float*)d_ws;
  float* attn_logits = ws;                   // 512
  float* attn_wt     = ws + 512;             // 512
  float* attn_app    = ws + 1024;            // 1024
  float* xcomb       = ws + 2048;            // 1024
  float* gi          = ws + 3072;            // 3072
  float* gh          = ws + 6144;            // 3072
  float* hnew        = ws + 9216;            // 1024
  _Float16* h16      = (_Float16*)(ws + 10240);  // 1024 halves (512 floats)
  float* logits      = ws + 10752;           // 50257
  float* red2        = ws + 61012;           // 2

  // k1: 512 rows, 1 wave per row
  k1_attn_logits<<<LSZ / 4, 128, 0, stream>>>(tok, hid, emb, attn_w, attn_b, attn_logits);
  // k2: softmax(512)
  k2_softmax<<<1, LSZ, 0, stream>>>(attn_logits, attn_wt, out_attn);
  // k3: weighted sum over encoder outputs
  k3_attn_applied<<<HSZ / 256, 256, 0, stream>>>(attn_wt, enc, attn_app);
  // k4: combine + relu, 1024 rows
  k4_combine<<<ESZ / 4, 128, 0, stream>>>(tok, emb, attn_app, comb_w, comb_b, xcomb);
  // k5: GRU gate GEMVs, 3072 rows
  k5_gates<<<(3 * HSZ) / 4, 128, 0, stream>>>(xcomb, hid, w_ih, w_hh, b_ih, b_hh, gi, gh);
  // k6: gate math -> h_new (f32 out + f16 staging)
  k6_hnew<<<HSZ / 256, 256, 0, stream>>>(gi, gh, hid, hnew, out_hnew, h16);
  // k7: big output projection via WMMA, 16 logits per wave
  {
    const int waves  = (VSZ + 15) / 16;       // 3142
    const int blocks = (waves + 3) / 4;       // 786 blocks of 4 waves
    k7_out_gemv<<<blocks, 128, 0, stream>>>(out_w, out_b, h16, logits);
  }
  // k8: max + log-sum-exp
  k8_lse<<<1, 256, 0, stream>>>(logits, red2);
  // k9: final log-softmax write
  k9_logsoftmax<<<(VSZ + 255) / 256, 256, 0, stream>>>(logits, red2, out);
}
